// STBLISModule_21646635171931
// MI455X (gfx1250) — matrix operand order repaired and checked by reading, not compiled, not run on hardware
//
#include <hip/hip_runtime.h>

// STBLIS forward for MI455X (gfx1250, wave32).
// y[b,c,p,q] = S_p @ X[b,c] @ T_q^T ; out = concat(relu(y), relu(-y)) on channels.
// Strategy: bf16 hi/lo split-precision (3x v_wmma_f32_16x16x32_bf16 per K-step),
// fp32 accumulation. Memory-bound design (~310 MB HBM traffic @ 23.3 TB/s).

#define Bn 4
#define Cn 4
#define Nn 512
#define Tn 256
#define Kw 4
#define Jw 4
#define BCn (Bn * Cn)

typedef __attribute__((ext_vector_type(16))) __bf16 v16bf;
typedef __attribute__((ext_vector_type(8)))  float  v8f;

union Frag16 {
    v16bf v;
    uint4 q[2];
};

__device__ __forceinline__ unsigned short bf16hi(float f) {
    unsigned u = __float_as_uint(f);
    unsigned r = u + 0x7FFFu + ((u >> 16) & 1u);   // round-to-nearest-even
    return (unsigned short)(r >> 16);
}
__device__ __forceinline__ float bf16f(unsigned short h) {
    return __uint_as_float(((unsigned)h) << 16);
}

// ---------------------------------------------------------------------------
// Prep 1: elementwise fp32 -> (bf16 hi, bf16 lo) split (for s_wav, t_wav)
// ---------------------------------------------------------------------------
__global__ __launch_bounds__(256) void split_f32_bf16x2(
    const float* __restrict__ src,
    unsigned short* __restrict__ hi,
    unsigned short* __restrict__ lo,
    int n)
{
    int i = blockIdx.x * 256 + threadIdx.x;
    if (i < n) {
        float v = src[i];
        unsigned short h = bf16hi(v);
        hi[i] = h;
        lo[i] = bf16hi(v - bf16f(h));
    }
}

// ---------------------------------------------------------------------------
// Prep 2: transpose X[b,c,m,u] -> Xt[b,c,u,m] with hi/lo bf16 split.
// LDS 32x32 tile, block (32,8), grid (N/32, T/32, B*C).
// ---------------------------------------------------------------------------
__global__ __launch_bounds__(256) void xpose_split(
    const float* __restrict__ x,
    unsigned short* __restrict__ xt_hi,
    unsigned short* __restrict__ xt_lo)
{
    __shared__ float tile[32][33];
    const int bc = blockIdx.z;
    const int m0 = blockIdx.x * 32;   // along N
    const int u0 = blockIdx.y * 32;   // along T
    const int tx = threadIdx.x, ty = threadIdx.y;

    const float* xp = x + (size_t)bc * Nn * Tn;
    for (int r = 0; r < 32; r += 8)
        tile[ty + r][tx] = xp[(size_t)(m0 + ty + r) * Tn + (u0 + tx)];
    __syncthreads();

    unsigned short* oh = xt_hi + (size_t)bc * Tn * Nn;
    unsigned short* ol = xt_lo + (size_t)bc * Tn * Nn;
    for (int r = 0; r < 32; r += 8) {
        float v = tile[tx][ty + r];                       // = x[m0+tx][u0+ty+r]
        size_t o = (size_t)(u0 + ty + r) * Nn + (m0 + tx);
        unsigned short h = bf16hi(v);
        oh[o] = h;
        ol[o] = bf16hi(v - bf16f(h));
    }
}

// ---------------------------------------------------------------------------
// Stage 1: tmp[bc,p] = S_p @ X[bc]   (M=N=512, K=512, cols=T=256), fp32 acc,
// stored as bf16 hi/lo. Wave tile 32x32 (2x2 WMMA accs), block 8 waves =
// 128x64 tile. Grid (4, 4, 64).
// ---------------------------------------------------------------------------
__global__ __launch_bounds__(256) void stage1_gemm(
    const unsigned short* __restrict__ s_hi, const unsigned short* __restrict__ s_lo,
    const unsigned short* __restrict__ xt_hi, const unsigned short* __restrict__ xt_lo,
    unsigned short* __restrict__ tmp_hi, unsigned short* __restrict__ tmp_lo)
{
    const int z    = blockIdx.z;        // bc*4 + p
    const int p    = z & 3;
    const int bc   = z >> 2;
    const int tid  = threadIdx.x;
    const int lane = tid & 31;
    const int l16  = lane & 15;
    const int half = lane >> 4;
    const int w    = tid >> 5;
    const int wr   = w >> 1;            // 0..3
    const int wc   = w & 1;             // 0..1
    const int row0 = blockIdx.x * 128 + wr * 32;
    const int col0 = blockIdx.y * 64  + wc * 32;

    v8f acc[2][2] = {};

    const uint4* a_hi[2]; const uint4* a_lo[2];
    const uint4* b_hi[2]; const uint4* b_lo[2];
    for (int mi = 0; mi < 2; ++mi) {
        size_t r = ((size_t)p * Nn + (row0 + mi * 16 + l16)) * Nn;
        a_hi[mi] = (const uint4*)(s_hi + r);
        a_lo[mi] = (const uint4*)(s_lo + r);
    }
    for (int ni = 0; ni < 2; ++ni) {
        size_t r = ((size_t)bc * Tn + (col0 + ni * 16 + l16)) * Nn;
        b_hi[ni] = (const uint4*)(xt_hi + r);
        b_lo[ni] = (const uint4*)(xt_lo + r);
    }
    // A frag: elem e -> k = (e&7) + 8*half + 16*(e>>3)  => two b128 chunks
    // B frag: elem e -> k = (e&7) + 8*half + 16*(e>>3)                 => two b128 chunks
    const int a0 = half;        // (8*half)/8
    const int a1 = 2 + half;    // (16+8*half)/8
    const int bq = half;        // (8*half)/8

    for (int k0 = 0; k0 < Nn; k0 += 32) {
        const int kq = k0 >> 3;
        Frag16 Ah[2], Al[2], Bh[2], Bl[2];
        for (int mi = 0; mi < 2; ++mi) {
            Ah[mi].q[0] = a_hi[mi][kq + a0]; Ah[mi].q[1] = a_hi[mi][kq + a1];
            Al[mi].q[0] = a_lo[mi][kq + a0]; Al[mi].q[1] = a_lo[mi][kq + a1];
        }
        for (int ni = 0; ni < 2; ++ni) {
            Bh[ni].q[0] = b_hi[ni][kq + bq]; Bh[ni].q[1] = b_hi[ni][kq + bq + 2];
            Bl[ni].q[0] = b_lo[ni][kq + bq]; Bl[ni].q[1] = b_lo[ni][kq + bq + 2];
        }
        for (int mi = 0; mi < 2; ++mi)
            for (int ni = 0; ni < 2; ++ni) {
                acc[mi][ni] = __builtin_amdgcn_wmma_f32_16x16x32_bf16(
                    false, Ah[mi].v, false, Bh[ni].v, (short)0, acc[mi][ni], false, false);
                acc[mi][ni] = __builtin_amdgcn_wmma_f32_16x16x32_bf16(
                    false, Ah[mi].v, false, Bl[ni].v, (short)0, acc[mi][ni], false, false);
                acc[mi][ni] = __builtin_amdgcn_wmma_f32_16x16x32_bf16(
                    false, Al[mi].v, false, Bh[ni].v, (short)0, acc[mi][ni], false, false);
            }
    }

    unsigned short* th = tmp_hi + (size_t)z * Nn * Tn;
    unsigned short* tl = tmp_lo + (size_t)z * Nn * Tn;
    for (int mi = 0; mi < 2; ++mi)
        for (int ni = 0; ni < 2; ++ni)
            for (int r = 0; r < 8; ++r) {
                int row = row0 + mi * 16 + r + 8 * half;   // C/D: M = r + 8*half
                int col = col0 + ni * 16 + l16;            // C/D: N = lane&15
                float v = acc[mi][ni][r];
                size_t o = (size_t)row * Tn + col;
                unsigned short h = bf16hi(v);
                th[o] = h;
                tl[o] = bf16hi(v - bf16f(h));
            }
}

// ---------------------------------------------------------------------------
// Stage 2: Y = tmp[bc,p] @ T_q^T  (M=512, K=256, cols=256); fused dual ReLU
// into out[b, ch, n, t] / out[b, ch+64, n, t]. Grid (4, 4, 256).
// B operand: B[k=u, col=t] = t_wav[q][t][u] -> rows contiguous in u.
// ---------------------------------------------------------------------------
__global__ __launch_bounds__(256) void stage2_gemm(
    const unsigned short* __restrict__ tmp_hi, const unsigned short* __restrict__ tmp_lo,
    const unsigned short* __restrict__ tw_hi, const unsigned short* __restrict__ tw_lo,
    float* __restrict__ out)
{
    const int z    = blockIdx.z;        // ((bc*4+p)*4)+q
    const int q    = z & 3;
    const int zt   = z >> 2;            // bc*4 + p
    const int p    = zt & 3;
    const int bc   = z >> 4;
    const int b    = bc >> 2;
    const int c    = bc & 3;
    const int tid  = threadIdx.x;
    const int lane = tid & 31;
    const int l16  = lane & 15;
    const int half = lane >> 4;
    const int w    = tid >> 5;
    const int wr   = w >> 1;
    const int wc   = w & 1;
    const int row0 = blockIdx.x * 128 + wr * 32;
    const int col0 = blockIdx.y * 64  + wc * 32;

    v8f acc[2][2] = {};

    const uint4* a_hi[2]; const uint4* a_lo[2];
    const uint4* b_hi[2]; const uint4* b_lo[2];
    for (int mi = 0; mi < 2; ++mi) {
        size_t r = ((size_t)zt * Nn + (row0 + mi * 16 + l16)) * Tn;
        a_hi[mi] = (const uint4*)(tmp_hi + r);
        a_lo[mi] = (const uint4*)(tmp_lo + r);
    }
    for (int ni = 0; ni < 2; ++ni) {
        size_t r = ((size_t)q * Tn + (col0 + ni * 16 + l16)) * Tn;
        b_hi[ni] = (const uint4*)(tw_hi + r);
        b_lo[ni] = (const uint4*)(tw_lo + r);
    }
    const int a0 = half;
    const int a1 = 2 + half;
    const int bq = half;

    for (int k0 = 0; k0 < Tn; k0 += 32) {
        const int kq = k0 >> 3;
        Frag16 Ah[2], Al[2], Bh[2], Bl[2];
        for (int mi = 0; mi < 2; ++mi) {
            Ah[mi].q[0] = a_hi[mi][kq + a0]; Ah[mi].q[1] = a_hi[mi][kq + a1];
            Al[mi].q[0] = a_lo[mi][kq + a0]; Al[mi].q[1] = a_lo[mi][kq + a1];
        }
        for (int ni = 0; ni < 2; ++ni) {
            Bh[ni].q[0] = b_hi[ni][kq + bq]; Bh[ni].q[1] = b_hi[ni][kq + bq + 2];
            Bl[ni].q[0] = b_lo[ni][kq + bq]; Bl[ni].q[1] = b_lo[ni][kq + bq + 2];
        }
        for (int mi = 0; mi < 2; ++mi)
            for (int ni = 0; ni < 2; ++ni) {
                acc[mi][ni] = __builtin_amdgcn_wmma_f32_16x16x32_bf16(
                    false, Ah[mi].v, false, Bh[ni].v, (short)0, acc[mi][ni], false, false);
                acc[mi][ni] = __builtin_amdgcn_wmma_f32_16x16x32_bf16(
                    false, Ah[mi].v, false, Bl[ni].v, (short)0, acc[mi][ni], false, false);
                acc[mi][ni] = __builtin_amdgcn_wmma_f32_16x16x32_bf16(
                    false, Al[mi].v, false, Bh[ni].v, (short)0, acc[mi][ni], false, false);
            }
    }

    const int ch = c * 16 + p * 4 + q;                       // d = p*J + q ordering
    const size_t chStride = (size_t)Nn * Tn;
    float* opos = out + ((size_t)b * (2 * Cn * Kw * Jw) + ch) * chStride;
    float* oneg = opos + (size_t)(Cn * Kw * Jw) * chStride;  // +64 channels

    for (int mi = 0; mi < 2; ++mi)
        for (int ni = 0; ni < 2; ++ni)
            for (int r = 0; r < 8; ++r) {
                int row = row0 + mi * 16 + r + 8 * half;
                int col = col0 + ni * 16 + l16;
                float v = acc[mi][ni][r];
                size_t o = (size_t)row * Tn + col;
                opos[o] = fmaxf(v, 0.0f);
                oneg[o] = fmaxf(-v, 0.0f);
            }
}

// ---------------------------------------------------------------------------
// Host launcher
// ---------------------------------------------------------------------------
extern "C" void kernel_launch(void* const* d_in, const int* in_sizes, int n_in,
                              void* d_out, int out_size, void* d_ws, size_t ws_size,
                              hipStream_t stream) {
    const float* x     = (const float*)d_in[0];   // [B,C,N,T]
    const float* s_wav = (const float*)d_in[1];   // [K,N,N]
    const float* t_wav = (const float*)d_in[2];   // [J,T,T]
    float* out = (float*)d_out;                   // [B, 2*C*K*J, N, T]

    const size_t nS  = (size_t)Kw * Nn * Nn;      // 1,048,576
    const size_t nT  = (size_t)Jw * Tn * Tn;      //   262,144
    const size_t nXt = (size_t)BCn * Tn * Nn;     // 2,097,152
    const size_t nTm = (size_t)BCn * Kw * Nn * Tn;// 8,388,608

    unsigned short* w = (unsigned short*)d_ws;
    unsigned short* s_hi   = w;            w += nS;
    unsigned short* s_lo   = w;            w += nS;
    unsigned short* tw_hi  = w;            w += nT;
    unsigned short* tw_lo  = w;            w += nT;
    unsigned short* xt_hi  = w;            w += nXt;
    unsigned short* xt_lo  = w;            w += nXt;
    unsigned short* tmp_hi = w;            w += nTm;
    unsigned short* tmp_lo = w;            w += nTm;
    // total: ~45 MB of d_ws

    split_f32_bf16x2<<<dim3((unsigned)((nS + 255) / 256)), dim3(256), 0, stream>>>(
        s_wav, s_hi, s_lo, (int)nS);
    split_f32_bf16x2<<<dim3((unsigned)((nT + 255) / 256)), dim3(256), 0, stream>>>(
        t_wav, tw_hi, tw_lo, (int)nT);
    xpose_split<<<dim3(Nn / 32, Tn / 32, BCn), dim3(32, 8), 0, stream>>>(
        x, xt_hi, xt_lo);

    stage1_gemm<<<dim3(Nn / 128, Tn / 64, BCn * Kw), dim3(256), 0, stream>>>(
        s_hi, s_lo, xt_hi, xt_lo, tmp_hi, tmp_lo);

    stage2_gemm<<<dim3(Nn / 128, Tn / 64, BCn * Kw * Jw), dim3(256), 0, stream>>>(
        tmp_hi, tmp_lo, tw_hi, tw_lo, out);
}
